// RNN_91302414778811
// MI455X (gfx1250) — matrix operand order repaired and checked by reading, not compiled, not run on hardware
//
#include <hip/hip_runtime.h>

// ---------------------------------------------------------------------------
// LSTM (B=32, T=4096, D=256, H=256) + FC head for MI455X (gfx1250, wave32)
//
// 16 persistent workgroups, each owning 16 hidden units (64 gate columns).
// W_lstm slice pre-packed in LDS as v_wmma_f32_16x16x32_bf16 B-fragments.
// Per timestep: TDM (tensor_load_to_lds) stages x_t (2-D strided) and h
// (1-D with LDS padding -> lands interleaved in the A staging buffer),
// 16 WMMAs per wave, f32 cell update with native v_tanh_f32 / v_exp_f32,
// h ping-pongs through L2 (bf16), agent-scope atomic grid barrier.
// ---------------------------------------------------------------------------

typedef __attribute__((ext_vector_type(16))) __bf16 v16bf;
typedef __attribute__((ext_vector_type(8)))  __bf16 v8bf;
typedef __attribute__((ext_vector_type(8)))  float  v8f;
typedef __attribute__((ext_vector_type(4)))  float  v4f;
typedef __attribute__((ext_vector_type(2)))  float  v2f;
typedef __attribute__((ext_vector_type(4)))  unsigned v4u;
typedef __attribute__((ext_vector_type(8)))  int      v8i;
typedef __attribute__((ext_vector_type(4)))  int      v4i;

#define BATCH   32
#define TSTEPS  4096
#define DIN     256
#define HID     256
#define KTOT    512      // DIN + HID
#define GCOLS   1024     // 4*HID
#define NWG     16
#define HORIZON 24

// workspace layout (bytes)
#define WS_BAR   0                        // u32 barrier counter
#define WS_HBUF  1024                     // bf16 [2][32][256] = 32768 B (ping-pong h)
#define WS_HFIN  (WS_HBUF + 32768)        // f32  [32][256]    = 32768 B (final h)
#define WS_FC1   (WS_HFIN + 32768)        // f32  [32][256]    = 32768 B (h @ W_fc)

// LDS layout (bytes)
#define LDS_XH     0                              // bf16 [32][512]      = 32768
#define LDS_XSTAGE 32768                          // f32  [32][256]      = 32768
#define LDS_WFRAG  (32768 + 32768)                // bf16 64 frags*32*16 = 65536
#define LDS_GATES  (LDS_WFRAG + 65536)            // f32 [4][32][16]     = 8192
#define LDS_CST    (LDS_GATES + 8192)             // f32 [32][16]        = 2048
#define LDS_TOTAL  (LDS_CST + 2048)

#if __has_builtin(__builtin_amdgcn_tensor_load_to_lds) && \
    __has_builtin(__builtin_amdgcn_s_wait_tensorcnt)
#define USE_TDM 1
#else
#define USE_TDM 0
#endif

// ---- fast transcendentals (native gfx1250 TRANS ops, guarded) ----
__device__ __forceinline__ float fast_exp(float x) {  // e^x
#if __has_builtin(__builtin_amdgcn_exp2f)
    return __builtin_amdgcn_exp2f(x * 1.442695041f);
#else
    return __expf(x);
#endif
}
__device__ __forceinline__ float fast_sigm(float x) {
#if __has_builtin(__builtin_amdgcn_rcpf)
    return __builtin_amdgcn_rcpf(1.0f + fast_exp(-x));
#else
    return 1.0f / (1.0f + fast_exp(-x));
#endif
}
__device__ __forceinline__ float fast_tanh(float x) {
#if __has_builtin(__builtin_amdgcn_tanhf)
    return __builtin_amdgcn_tanhf(x);   // v_tanh_f32
#else
    float e = fast_exp(2.0f * x);
    return 1.0f - 2.0f / (e + 1.0f);
#endif
}

__device__ __forceinline__ void grid_barrier(unsigned* bar, unsigned nbar) {
    __syncthreads();
    if (threadIdx.x == 0) {
        __threadfence();  // make h stores visible device-wide before arrive
        __hip_atomic_fetch_add(bar, 1u, __ATOMIC_RELEASE, __HIP_MEMORY_SCOPE_AGENT);
        const unsigned target = nbar * NWG;
        while (__hip_atomic_load(bar, __ATOMIC_ACQUIRE, __HIP_MEMORY_SCOPE_AGENT) < target)
            __builtin_amdgcn_s_sleep(1);
    }
    __syncthreads();
}

#if USE_TDM
// Build D# group0: count=1, lds_addr, 57-bit global addr, type=2 (image)
__device__ __forceinline__ v4u tdm_group0(unsigned lds_addr, unsigned long long gaddr) {
    v4u g;
    g[0] = 1u;                                        // count=1, user mode
    g[1] = lds_addr;                                  // LDS byte address
    g[2] = (unsigned)(gaddr & 0xFFFFFFFFull);         // global_addr[31:0]
    g[3] = (unsigned)((gaddr >> 32) & 0x1FFFFFFull)   // global_addr[56:32]
         | (2u << 30);                                // type = 2
    return g;
}
#endif

__global__ void lstm_init_ws(void* ws) {
    // zero barrier counter + both h ping-pong buffers (bf16 zeros == 0x0000)
    unsigned* p = (unsigned*)ws;
    const unsigned nwords = (WS_HBUF + 32768) / 4;
    for (unsigned i = blockIdx.x * blockDim.x + threadIdx.x; i < nwords;
         i += gridDim.x * blockDim.x)
        p[i] = 0u;
}

__global__ __launch_bounds__(256) void lstm_persistent(
    const float* __restrict__ x,       // [32,4096,256]
    const float* __restrict__ W_lstm,  // [512,1024]
    const float* __restrict__ b_lstm,  // [1024]
    const float* __restrict__ W_fc,    // [256,256]
    const float* __restrict__ b_fc,    // [256]
    const float* __restrict__ W_out,   // [256,24]
    const float* __restrict__ b_out,   // [24]
    float* __restrict__ out,           // [32,24]
    void* __restrict__ ws)
{
    extern __shared__ char smem[];
    __bf16* xh     = (__bf16*)(smem + LDS_XH);      // [32][512] A-matrix staging
    float*  xstage = (float*)(smem + LDS_XSTAGE);   // [32][256] raw x_t tile
    __bf16* wfrag  = (__bf16*)(smem + LDS_WFRAG);   // packed B fragments
    float*  gates  = (float*)(smem + LDS_GATES);    // [4][32][16]
    float*  cst    = (float*)(smem + LDS_CST);      // [32][16] cell state

    const int tid  = threadIdx.x;
    const int lane = tid & 31;
    const int wave = tid >> 5;       // 8 waves
    const int wg   = blockIdx.x;     // 16 workgroups

    unsigned* bar  = (unsigned*)((char*)ws + WS_BAR);
    __bf16*   hbuf = (__bf16*)((char*)ws + WS_HBUF);  // [2][32][256]
    float*    hfin = (float*)((char*)ws + WS_HFIN);   // [32][256]
    float*    fc1  = (float*)((char*)ws + WS_FC1);    // [32][256]

    // ---- init cell state ----
    for (int i = tid; i < BATCH * 16; i += 256) cst[i] = 0.0f;

    // ---- pre-pack W_lstm slice into WMMA bf16 B-fragment layout ----
    {
        const int n     = lane & 15;
        const int khalf = lane >> 4;
        for (int fi = wave * 8; fi < wave * 8 + 8; ++fi) {
            const int ntile = fi >> 4;
            const int kt    = fi & 15;
            const int col   = ntile * HID + wg * 16 + n;  // global gate column
            __bf16* dst = wfrag + ((size_t)fi * 32 + lane) * 16;
#pragma unroll
            for (int e = 0; e < 16; ++e) {
                const int K = kt * 32 + khalf * 16 + e;
                dst[e] = (__bf16)W_lstm[(size_t)K * GCOLS + col];
            }
        }
    }

    // ---- per-wave constants for the gate GEMM ----
    const int   mtile = wave & 1;    // batch tile 0..1
    const int   ntile = wave >> 1;   // gate 0..3
    const int   wn    = lane & 15;
    const int   whalf = lane >> 4;
    const float biasv = b_lstm[ntile * HID + wg * 16 + wn];

#if USE_TDM
    // ---- loop-invariant TDM D# group1 descriptors ----
    // x_t tile: 2-D, 4-byte elems, tile 256 x 32, row stride T*D elements
    v8i g1x;
    g1x[0] = (int)(2u << 16);                 // data_size=4B
    g1x[1] = (int)((unsigned)DIN << 16);      // tensor_dim0[15:0]=256
    g1x[2] = (int)((unsigned)BATCH << 16);    // tensor_dim1[15:0]=32
    g1x[3] = (int)((unsigned)DIN << 16);      // tile_dim0=256
    g1x[4] = (int)BATCH;                      // tile_dim1=32
    g1x[5] = (int)(TSTEPS * DIN);             // tensor_dim0_stride[31:0]=1048576
    g1x[6] = 0;                               // stride[47:32]=0, dim1_stride lo=0
    g1x[7] = 0;
    // h tile: 1-D, 4-byte elems, 4096 DWORDs, LDS padding 128 DW every 128 DW
    // -> deposits 512B h-rows into cols 256..511 of each 1024B xh row
    v8i g1h;
    g1h[0] = (int)((2u << 16) | (1u << 20)    // data_size=4B, pad_enable
                 | (6u << 22)                 // pad_interval: 128 DWORDs
                 | (127u << 25));             // pad_amount:   128 DWORDs
    g1h[1] = (int)(4096u << 16);              // tensor_dim0=4096 DWORDs
    g1h[2] = (int)(1u << 16);                 // tensor_dim1=1
    g1h[3] = (int)(4096u << 16);              // tile_dim0=4096
    g1h[4] = 0;
    g1h[5] = 4096;                            // tensor_dim0_stride
    g1h[6] = 0;
    g1h[7] = 0;
    const v4i gz  = {0, 0, 0, 0};             // groups 2/3 unused (<=2-D)
    const v8i gz8 = {0, 0, 0, 0, 0, 0, 0, 0}; // extra descriptor group (clang-23 6-arg form)
    const unsigned lds_xstage = (unsigned)(uintptr_t)(void*)xstage;
    const unsigned lds_hdst   = (unsigned)(uintptr_t)(void*)(xh + HID); // row0 col256
#endif

    __syncthreads();
    unsigned nbar = 0;

    // ======================= sequential recurrence =======================
    for (int t = 0; t < TSTEPS; ++t) {
        // ---- stage x_t (raw f32) and h_{t-1} (bf16, padded deposit) ----
#if USE_TDM
        if (wave == 0) {
            const unsigned long long gx =
                (unsigned long long)(uintptr_t)(x + (size_t)t * DIN);
            const unsigned long long gh =
                (unsigned long long)(uintptr_t)(hbuf + (size_t)(t & 1) * BATCH * HID);
            __builtin_amdgcn_tensor_load_to_lds(tdm_group0(lds_xstage, gx), g1x, gz, gz, gz8, 0);
            __builtin_amdgcn_tensor_load_to_lds(tdm_group0(lds_hdst,   gh), g1h, gz, gz, gz8, 0);
            __builtin_amdgcn_s_wait_tensorcnt(0);
        }
        __syncthreads();
        // convert x_t tile f32 -> bf16 into xh cols 0..255
        {
            const int b   = tid >> 3;
            const int seg = tid & 7;
            const float* xs = xstage + b * DIN + seg * 32;
            __bf16* xd = xh + b * KTOT + seg * 32;
#pragma unroll
            for (int j = 0; j < 8; ++j) {
                v4f v = *(const v4f*)(xs + j * 4);
                xd[j * 4 + 0] = (__bf16)v[0];
                xd[j * 4 + 1] = (__bf16)v[1];
                xd[j * 4 + 2] = (__bf16)v[2];
                xd[j * 4 + 3] = (__bf16)v[3];
            }
            if (t + 1 < TSTEPS)  // warm L2 for next step's TDM fetch
                __builtin_prefetch(x + ((size_t)b * TSTEPS + t + 1) * DIN + seg * 32, 0, 0);
        }
#else
        {
            const int b   = tid >> 3;
            const int seg = tid & 7;
            const float* xrow = x + ((size_t)b * TSTEPS + t) * DIN + seg * 32;
            __bf16* xd = xh + b * KTOT + seg * 32;
#pragma unroll
            for (int j = 0; j < 8; ++j) {
                v4f v = *(const v4f*)(xrow + j * 4);
                xd[j * 4 + 0] = (__bf16)v[0];
                xd[j * 4 + 1] = (__bf16)v[1];
                xd[j * 4 + 2] = (__bf16)v[2];
                xd[j * 4 + 3] = (__bf16)v[3];
            }
            if (t + 1 < TSTEPS) __builtin_prefetch(xrow + DIN, 0, 0);
            const __bf16* hsrc = hbuf + (size_t)(t & 1) * BATCH * HID + b * HID + seg * 32;
            __bf16* hd = xh + b * KTOT + HID + seg * 32;
            *(v16bf*)(hd +  0) = *(const v16bf*)(hsrc +  0);
            *(v16bf*)(hd + 16) = *(const v16bf*)(hsrc + 16);
        }
#endif
        __syncthreads();

        // ---- gate GEMM: 16x16 D-tile per wave, K = 512 in 16 WMMAs ----
        v8f acc;
#pragma unroll
        for (int r = 0; r < 8; ++r) acc[r] = biasv;

        const int am = mtile * 16 + wn;   // A row for this lane
#pragma unroll 4
        for (int kt = 0; kt < 16; ++kt) {
            const int k0 = kt * 32 + whalf * 16;
            v8bf alo = *(const v8bf*)(xh + am * KTOT + k0);
            v8bf ahi = *(const v8bf*)(xh + am * KTOT + k0 + 8);
            v16bf A = __builtin_shufflevector(alo, ahi,
                          0, 1, 2, 3, 4, 5, 6, 7, 8, 9, 10, 11, 12, 13, 14, 15);
            v16bf Bm = *(const v16bf*)(wfrag + ((size_t)(ntile * 16 + kt) * 32 + lane) * 16);
            acc = __builtin_amdgcn_wmma_f32_16x16x32_bf16(
                      false, A, false, Bm, (short)0, acc, false, false);
        }

        // ---- scatter gates to LDS: [gate][batch][unit] ----
#pragma unroll
        for (int r = 0; r < 8; ++r) {
            const int m = mtile * 16 + whalf * 8 + r;  // batch row
            gates[(ntile * BATCH + m) * 16 + wn] = acc[r];
        }
        __syncthreads();

        // ---- cell update: 512 (b,u) pairs, 2 per thread, f32 ----
#pragma unroll
        for (int q = 0; q < 2; ++q) {
            const int p = tid * 2 + q;
            const int b = p >> 4;
            const int u = p & 15;
            const float gi = gates[(0 * BATCH + b) * 16 + u];
            const float gg = gates[(1 * BATCH + b) * 16 + u];
            const float gf = gates[(2 * BATCH + b) * 16 + u];
            const float go = gates[(3 * BATCH + b) * 16 + u];
            float c = cst[b * 16 + u];
            c = fast_sigm(gf + 1.0f) * c + fast_sigm(gi) * fast_tanh(gg);
            const float h = fast_sigm(go) * fast_tanh(c);
            cst[b * 16 + u] = c;
            const int col = wg * 16 + u;
            hbuf[(size_t)((t + 1) & 1) * BATCH * HID + b * HID + col] = (__bf16)h;
            if (t == TSTEPS - 1) hfin[b * HID + col] = h;
        }

        grid_barrier(bar, ++nbar);
    }

    // ======================= fc1 = h @ W_fc + b_fc (f32 WMMA) ===========
    if (wave < 2) {
        const int fm   = wave;               // M tile
        const int n    = lane & 15;
        const int kh   = lane >> 4;
        const int col  = wg * 16 + n;
        const float bb = b_fc[col];
        v8f facc;
#pragma unroll
        for (int r = 0; r < 8; ++r) facc[r] = bb;

        const int m = fm * 16 + n;
        for (int kt = 0; kt < HID / 4; ++kt) {
            const int kb = kt * 4 + kh * 2;
            v2f a, bv;
            a[0]  = hfin[m * HID + kb + 0];
            a[1]  = hfin[m * HID + kb + 1];
            bv[0] = W_fc[(size_t)(kb + 0) * HID + col];
            bv[1] = W_fc[(size_t)(kb + 1) * HID + col];
            facc = __builtin_amdgcn_wmma_f32_16x16x4_f32(
                       false, a, false, bv, (short)0, facc, false, false);
        }
#pragma unroll
        for (int r = 0; r < 8; ++r) {
            const int mm = fm * 16 + kh * 8 + r;
            fc1[mm * HID + col] = facc[r];
        }
    }
    grid_barrier(bar, ++nbar);

    // ======================= out = fc1 @ W_out + b_out (tiny) ===========
    if (wg == 0) {
        for (int p = tid; p < BATCH * HORIZON; p += 256) {
            const int b = p / HORIZON;
            const int j = p % HORIZON;
            float s = b_out[j];
            for (int k = 0; k < HID; ++k)
                s += fc1[b * HID + k] * W_out[k * HORIZON + j];
            out[p] = s;
        }
    }
}

extern "C" void kernel_launch(void* const* d_in, const int* in_sizes, int n_in,
                              void* d_out, int out_size, void* d_ws, size_t ws_size,
                              hipStream_t stream) {
    const float* x      = (const float*)d_in[0];
    const float* W_lstm = (const float*)d_in[1];
    const float* b_lstm = (const float*)d_in[2];
    const float* W_fc   = (const float*)d_in[3];
    const float* b_fc   = (const float*)d_in[4];
    const float* W_out  = (const float*)d_in[5];
    const float* b_out  = (const float*)d_in[6];
    float* out = (float*)d_out;

    lstm_init_ws<<<40, 256, 0, stream>>>(d_ws);
    lstm_persistent<<<NWG, 256, LDS_TOTAL, stream>>>(
        x, W_lstm, b_lstm, W_fc, b_fc, W_out, b_out, out, d_ws);
}
